// NerfNGP_7327214207035
// MI455X (gfx1250) — compile-verified, matrix operand order checked by reading, not run on hardware
//
#include <hip/hip_runtime.h>
#include <hip/hip_bf16.h>

#define NTHREADS 256
#define ROWS_PER_BLK 128
#define T_SIZE (1 << 19)

typedef _Float16 v16h __attribute__((ext_vector_type(16)));
typedef _Float16 h8v  __attribute__((ext_vector_type(8)));
typedef float    v8f  __attribute__((ext_vector_type(8)));

// Load a 16x32 f16 fragment (A layout) from LDS, row-major with `stride` f16/row.
// Same loader serves B fragments when weights are stored transposed (W^T[out][in]):
// B column n == row n of W^T, identical K packing.
// CDNA5 16-bit A layout: lanes 0-15 -> K {0..7,16..23}; lanes 16-31 -> K {8..15,24..31}.
__device__ __forceinline__ v16h ld_frag(const _Float16* __restrict__ base, int stride, int lane) {
    const int r  = lane & 15;
    const int kb = (lane & 16) ? 8 : 0;
    const _Float16* p = base + r * stride + kb;
    h8v lo = *(const h8v*)(p);        // K = kb .. kb+7
    h8v hi = *(const h8v*)(p + 16);   // K = kb+16 .. kb+23
    v16h out;
#pragma unroll
    for (int i = 0; i < 8; ++i) { out[i] = lo[i]; out[i + 8] = hi[i]; }
    return out;
}

__device__ __forceinline__ v8f wmma_f16f32(v16h a, v16h b, v8f c) {
    return __builtin_amdgcn_wmma_f32_16x16x32_f16(false, a, false, b, (short)0, c, false, false);
}

// Write a 16x16 f32 D tile (C/D layout: lane -> col n=lane&15, rows (lane>>4)*8 .. +7)
// to f16 LDS with bias + ReLU.
__device__ __forceinline__ void st_tile_relu(v8f d, _Float16* __restrict__ dst, int stride,
                                             int ncol0, const float* __restrict__ bias, int lane) {
    const int n  = lane & 15;
    const int m0 = (lane >> 4) * 8;
    const float bb = bias[n];
#pragma unroll
    for (int v = 0; v < 8; ++v) {
        float x = d[v] + bb;
        x = fmaxf(x, 0.f);
        dst[(m0 + v) * stride + ncol0 + n] = (_Float16)x;
    }
}

__global__ __launch_bounds__(NTHREADS)
void nerf_ngp_fused(const float* __restrict__ position, const float* __restrict__ direction,
                    const float* __restrict__ tables,
                    const float* __restrict__ Wd1, const float* __restrict__ bd1,
                    const float* __restrict__ Wd2, const float* __restrict__ bd2,
                    const float* __restrict__ Wc1, const float* __restrict__ bc1,
                    const float* __restrict__ Wc2, const float* __restrict__ bc2,
                    const float* __restrict__ Wc3, const float* __restrict__ bc3,
                    float* __restrict__ rgb_out, float* __restrict__ sigma_out, int N)
{
    // ---- LDS: transposed f16 weights + staging + activation buffers (~66 KB) ----
    __shared__ __align__(16) _Float16 s_wd1[64 * 32];   // W_d1^T [out64][in32]
    __shared__ __align__(16) _Float16 s_wd2[16 * 64];   // W_d2^T [out16][in64]
    __shared__ __align__(16) _Float16 s_wc1[64 * 64];   // W_c1^T [out64][in64], K 41..63 zero
    __shared__ __align__(16) _Float16 s_wc2[64 * 64];   // W_c2^T [out64][in64]
    __shared__ __align__(16) _Float16 s_wc3[16 * 64];   // W_c3^T [out16][in64], rows 3..15 zero
    __shared__ float s_b[224];                          // bd1[0..63] bd2[64..79] bc1[80..143] bc2[144..207] bc3[208..223]
    __shared__ float s_pos[ROWS_PER_BLK * 3];
    __shared__ __align__(16) _Float16 s_enc [ROWS_PER_BLK * 32]; // hash features (MLP input A)
    __shared__ __align__(16) _Float16 s_h   [ROWS_PER_BLK * 64]; // hidden activations
    __shared__ __align__(16) _Float16 s_c1in[ROWS_PER_BLK * 64]; // [density16 | SH25 | 0-pad]

    const int tid = threadIdx.x;
    const int g0  = blockIdx.x * ROWS_PER_BLK;

    // ---- Phase A0: zero padded weight regions ----
    for (int i = tid; i < 64 * 64; i += NTHREADS) s_wc1[i] = (_Float16)0.f;
    for (int i = tid; i < 16 * 64; i += NTHREADS) s_wc3[i] = (_Float16)0.f;
    if (tid < 16) s_b[208 + tid] = 0.f;
    __syncthreads();

    // ---- Phase A1: weights -> f16 transposed LDS; biases; stage positions ----
    for (int e = tid; e < 32 * 64; e += NTHREADS) { int i = e >> 6, o = e & 63; s_wd1[o * 32 + i] = (_Float16)Wd1[e]; }
    for (int e = tid; e < 64 * 16; e += NTHREADS) { int i = e >> 4, o = e & 15; s_wd2[o * 64 + i] = (_Float16)Wd2[e]; }
    for (int e = tid; e < 41 * 64; e += NTHREADS) { int i = e >> 6, o = e & 63; s_wc1[o * 64 + i] = (_Float16)Wc1[e]; }
    for (int e = tid; e < 64 * 64; e += NTHREADS) { int i = e >> 6, o = e & 63; s_wc2[o * 64 + i] = (_Float16)Wc2[e]; }
    for (int e = tid; e < 64 * 3;  e += NTHREADS) { int i = e / 3,  o = e % 3;  s_wc3[o * 64 + i] = (_Float16)Wc3[e]; }
    if (tid < 64) s_b[tid]       = bd1[tid];
    if (tid < 16) s_b[64 + tid]  = bd2[tid];
    if (tid < 64) s_b[80 + tid]  = bc1[tid];
    if (tid < 64) s_b[144 + tid] = bc2[tid];
    if (tid < 3)  s_b[208 + tid] = bc3[tid];
    for (int i = tid; i < ROWS_PER_BLK * 3; i += NTHREADS) {
        long gi = (long)g0 * 3 + i;
        s_pos[i] = (gi < (long)N * 3) ? position[gi] : 0.f;
    }
    __syncthreads();

    // ---- Phase B: hash-grid encode (8 row/level tasks per thread) + SH encode ----
#pragma unroll
    for (int t = 0; t < 8; ++t) {
        const int e   = t * NTHREADS + tid;   // 0..2047
        const int row = e >> 4;
        const int lvl = e & 15;
        const float px = s_pos[row * 3 + 0], py = s_pos[row * 3 + 1], pz = s_pos[row * 3 + 2];
        const float res = (float)(16 << lvl); // b == 2 exactly for L=16, base 16 -> finest 16*2^15
        const float fx = px * res, fy = py * res, fz = pz * res;
        const float bx = floorf(fx), by = floorf(fy), bz = floorf(fz);
        const float wx = fx - bx, wy = fy - by, wz = fz - bz;
        const unsigned ix = (unsigned)(int)bx, iy = (unsigned)(int)by, iz = (unsigned)(int)bz;
        const float2* tab = (const float2*)tables + (size_t)lvl * T_SIZE;
        float f0 = 0.f, f1 = 0.f;
#pragma unroll
        for (int c = 0; c < 8; ++c) {
            const unsigned i = (c >> 2) & 1u, j = (c >> 1) & 1u, k = c & 1u;
            unsigned h = ((ix + i) * 1u) ^ ((iy + j) * 2654435761u) ^ ((iz + k) * 805459861u);
            h &= (T_SIZE - 1);
            const float2 emb = tab[h];
            const float cw = (i ? wx : 1.f - wx) * (j ? wy : 1.f - wy) * (k ? wz : 1.f - wz);
            f0 += cw * emb.x;
            f1 += cw * emb.y;
        }
        s_enc[row * 32 + lvl * 2 + 0] = (_Float16)f0;
        s_enc[row * 32 + lvl * 2 + 1] = (_Float16)f1;
    }

    if (tid < ROWS_PER_BLK) {
        const int row = tid;
        long gr = (long)(g0 + row);
        if (gr >= N) gr = N - 1;
        const float x = direction[gr * 3 + 0], y = direction[gr * 3 + 1], z = direction[gr * 3 + 2];
        const float x2 = x * x, y2 = y * y, z2 = z * z;
        const float xy = x * y, xz = x * z, yz = y * z;
        const float x4 = x2 * x2, y4 = y2 * y2;
        float sh[25];
        sh[0]  = 0.28209479177387814f;
        sh[1]  = -0.4886025119029199f * y;
        sh[2]  =  0.4886025119029199f * z;
        sh[3]  = -0.4886025119029199f * x;
        sh[4]  =  1.0925484305920792f * xy;
        sh[5]  = -1.0925484305920792f * yz;
        sh[6]  =  0.9461746957575601f * z2 - 0.31539156525252005f;
        sh[7]  = -1.0925484305920792f * xz;
        sh[8]  =  0.5462742152960396f * (x2 - y2);
        sh[9]  = -0.5900435899266435f * y * (3.f * x2 - y2);
        sh[10] =  2.890611442640554f * xy * z;
        sh[11] =  0.30469719964f * y * (1.5f - 7.5f * z2);
        sh[12] =  1.24392110863372f * z * (1.5f * z2 - 0.5f) - 0.497568443453487f * z;
        sh[13] =  0.30469719964f * x * (1.5f - 7.5f * z2);
        sh[14] =  1.445305721320277f * z * (x2 - y2);
        sh[15] = -0.5900435899266435f * x * (x2 - 3.f * y2);
        sh[16] =  2.5033429417967f * xy * (x2 - y2);
        sh[17] = -1.77013076977993f * yz * (3.f * x2 - y2);
        sh[18] =  0.126156626101008f * xy * (52.5f * z2 - 7.5f);
        sh[19] =  0.267618617422916f * y * (2.33333333333333f * z * (1.5f - 7.5f * z2) + 4.f * z);
        sh[20] =  1.48099765681286f * z * (1.66666666666667f * z * (1.5f * z2 - 0.5f) - 0.666666666666667f * z)
                 - 0.952069922236839f * z2 + 0.317356640745613f;
        sh[21] =  0.267618617422916f * x * (2.33333333333333f * z * (1.5f - 7.5f * z2) + 4.f * z);
        sh[22] =  0.063078313050504f * (x2 - y2) * (52.5f * z2 - 7.5f);
        sh[23] = -1.77013076977993f * xz * (x2 - 3.f * y2);
        sh[24] = -3.75501441269506f * x2 * y2 + 0.625835735449176f * x4 + 0.625835735449176f * y4;
        _Float16* dst = &s_c1in[row * 64 + 16];
#pragma unroll
        for (int j = 0; j < 25; ++j) dst[j] = (_Float16)sh[j];
#pragma unroll
        for (int j = 25; j < 48; ++j) dst[j] = (_Float16)0.f;
    }
    __syncthreads();

    // ---- Phase C: fully-fused MLP, one 16-row tile per wave (24 WMMAs/wave) ----
    const int lane = tid & 31;
    const int wv   = tid >> 5;
    const int r0   = wv * 16;
    const v8f zero8 = {0.f, 0.f, 0.f, 0.f, 0.f, 0.f, 0.f, 0.f};

    // density layer 1: [16x32] x [32x64] -> relu -> s_h
    {
        const v16h a = ld_frag(&s_enc[r0 * 32], 32, lane);
#pragma unroll
        for (int nt = 0; nt < 4; ++nt) {
            const v16h b = ld_frag(&s_wd1[(nt * 16) * 32], 32, lane);
            v8f d = wmma_f16f32(a, b, zero8);
            st_tile_relu(d, &s_h[r0 * 64], 64, nt * 16, &s_b[0 + nt * 16], lane);
        }
    }

    // density layer 2: [16x64] x [64x16] -> density_out (no relu); sigma = relu(col 15)
    {
        v8f d = zero8;
#pragma unroll
        for (int kt = 0; kt < 2; ++kt) {
            const v16h a = ld_frag(&s_h[r0 * 64 + kt * 32], 64, lane);
            const v16h b = ld_frag(&s_wd2[kt * 32], 64, lane);
            d = wmma_f16f32(a, b, d);
        }
        const int n = lane & 15, m0 = (lane >> 4) * 8;
        const float bb = s_b[64 + n];
        _Float16* dst = &s_c1in[r0 * 64];
#pragma unroll
        for (int v = 0; v < 8; ++v) {
            const float val = d[v] + bb;
            dst[(m0 + v) * 64 + n] = (_Float16)val;
            const long grow = (long)(g0 + r0 + m0 + v);
            if (n == 15 && grow < N) sigma_out[grow] = fmaxf(val, 0.f);
        }
    }

    // color layer 1: [16x64(density|SH|0)] x [64x64] -> relu -> s_h
#pragma unroll
    for (int nt = 0; nt < 4; ++nt) {
        v8f d = zero8;
#pragma unroll
        for (int kt = 0; kt < 2; ++kt) {
            const v16h a = ld_frag(&s_c1in[r0 * 64 + kt * 32], 64, lane);
            const v16h b = ld_frag(&s_wc1[(nt * 16) * 64 + kt * 32], 64, lane);
            d = wmma_f16f32(a, b, d);
        }
        st_tile_relu(d, &s_h[r0 * 64], 64, nt * 16, &s_b[80 + nt * 16], lane);
    }

    // color layer 2: [16x64] x [64x64] -> relu -> s_c1in (reuse)
#pragma unroll
    for (int nt = 0; nt < 4; ++nt) {
        v8f d = zero8;
#pragma unroll
        for (int kt = 0; kt < 2; ++kt) {
            const v16h a = ld_frag(&s_h[r0 * 64 + kt * 32], 64, lane);
            const v16h b = ld_frag(&s_wc2[(nt * 16) * 64 + kt * 32], 64, lane);
            d = wmma_f16f32(a, b, d);
        }
        st_tile_relu(d, &s_c1in[r0 * 64], 64, nt * 16, &s_b[144 + nt * 16], lane);
    }

    // color layer 3: [16x64] x [64x16(pad of 64x3)] -> sigmoid -> rgb
    {
        v8f d = zero8;
#pragma unroll
        for (int kt = 0; kt < 2; ++kt) {
            const v16h a = ld_frag(&s_c1in[r0 * 64 + kt * 32], 64, lane);
            const v16h b = ld_frag(&s_wc3[kt * 32], 64, lane);
            d = wmma_f16f32(a, b, d);
        }
        const int n = lane & 15, m0 = (lane >> 4) * 8;
        if (n < 3) {
            const float bb = s_b[208 + n];
#pragma unroll
            for (int v = 0; v < 8; ++v) {
                const long grow = (long)(g0 + r0 + m0 + v);
                if (grow < N) {
                    const float x = d[v] + bb;
                    rgb_out[grow * 3 + n] = 1.f / (1.f + __expf(-x));
                }
            }
        }
    }
}

extern "C" void kernel_launch(void* const* d_in, const int* in_sizes, int n_in,
                              void* d_out, int out_size, void* d_ws, size_t ws_size,
                              hipStream_t stream) {
    (void)n_in; (void)out_size; (void)d_ws; (void)ws_size;
    const float* position  = (const float*)d_in[0];
    const float* direction = (const float*)d_in[1];
    const float* tables    = (const float*)d_in[2];
    const float* Wd1 = (const float*)d_in[3];
    const float* bd1 = (const float*)d_in[4];
    const float* Wd2 = (const float*)d_in[5];
    const float* bd2 = (const float*)d_in[6];
    const float* Wc1 = (const float*)d_in[7];
    const float* bc1 = (const float*)d_in[8];
    const float* Wc2 = (const float*)d_in[9];
    const float* bc2 = (const float*)d_in[10];
    const float* Wc3 = (const float*)d_in[11];
    const float* bc3 = (const float*)d_in[12];

    const int N = in_sizes[0] / 3;
    float* rgb   = (float*)d_out;           // [N,3]
    float* sigma = rgb + (size_t)N * 3;     // [N,1]

    const int blocks = (N + ROWS_PER_BLK - 1) / ROWS_PER_BLK;
    hipLaunchKernelGGL(nerf_ngp_fused, dim3(blocks), dim3(NTHREADS), 0, stream,
                       position, direction, tables,
                       Wd1, bd1, Wd2, bd2, Wc1, bc1, Wc2, bc2, Wc3, bc3,
                       rgb, sigma, N);
}
